// GraphAttention_14740327760460
// MI455X (gfx1250) — matrix operand order repaired and checked by reading, not compiled, not run on hardware
//
#include <hip/hip_runtime.h>
#include <hip/hip_bf16.h>

// GAT forward, MI455X / gfx1250, wave32 + WMMA bf16 + adj bitmask.
// N=4096, F_IN=128, F_HID=64, H=8.
#define NN    4096
#define NW    (NN / 32)   // 128 mask words per row
#define FIN   128
#define FHID  64
#define HH    8
#define ALPHA 0.2f

typedef __bf16 bf16_t;
typedef bf16_t v16bf __attribute__((ext_vector_type(16)));
typedef float  v8f   __attribute__((ext_vector_type(8)));

__device__ __forceinline__ float lrelu(float x) { return x > 0.f ? x : ALPHA * x; }
__device__ __forceinline__ float rdlane(float v, int l) {
  return __int_as_float(__builtin_amdgcn_readlane(__float_as_int(v), l));
}

// ---------------- prep: adj -> bitmask (wave32 ballot) ----------------
__global__ void k_mask(const float* __restrict__ adj, unsigned* __restrict__ mask) {
  size_t idx = (size_t)blockIdx.x * blockDim.x + threadIdx.x;
  float  av  = adj[idx];
  unsigned long long b = __ballot(av > 0.f);
  if ((threadIdx.x & 31) == 0) mask[idx >> 5] = (unsigned)b;
}

// ---------------- prep: f32 -> bf16 conversions ----------------
__global__ void k_cvt_x(const float* __restrict__ x, bf16_t* __restrict__ xbf) {
  int i = blockIdx.x * blockDim.x + threadIdx.x;
  if (i < NN * FIN) xbf[i] = (bf16_t)x[i];
}
__global__ void k_cvt_w(const float* __restrict__ w, bf16_t* __restrict__ wbf) {
  int i = blockIdx.x * blockDim.x + threadIdx.x;
  if (i < HH * FIN * FHID) wbf[i] = (bf16_t)w[i];
}

// ---------------- Wh = x @ W_heads (per head), bf16 WMMA ----------------
__global__ void __launch_bounds__(256) k_wh(const bf16_t* __restrict__ xbf,
                                            const bf16_t* __restrict__ wbf,
                                            bf16_t* __restrict__ whp) {
  const int lane = threadIdx.x & 31;
  const int wid  = threadIdx.x >> 5;
  const int h    = blockIdx.y;
  const int rt   = blockIdx.x * 8 + wid;
  const int m    = lane & 15;
  const int kh   = lane >> 4;

  v8f acc[4];
#pragma unroll
  for (int dt = 0; dt < 4; ++dt)
#pragma unroll
    for (int q = 0; q < 8; ++q) acc[dt][q] = 0.f;

  const bf16_t* wh  = wbf + h * FIN * FHID;
  const int     row = rt * 16 + m;

  for (int kc = 0; kc < 4; ++kc) {
    v16bf a;
    {
      const int     k0 = kc * 32 + kh * 16;
      const bf16_t* xr = xbf + row * FIN + k0;
#pragma unroll
      for (int i = 0; i < 8; ++i) a[i] = xr[i];
#pragma unroll
      for (int i = 0; i < 8; ++i) a[8 + i] = xr[8 + i];
    }
#pragma unroll
    for (int dt = 0; dt < 4; ++dt) {
      v16bf b;
      const int d  = dt * 16 + m;
      const int kb = kc * 32 + kh * 16;
#pragma unroll
      for (int t = 0; t < 16; ++t) b[t] = wh[(kb + t) * FHID + d];
      acc[dt] = __builtin_amdgcn_wmma_f32_16x16x32_bf16(false, a, false, b,
                                                        (short)0, acc[dt], false, false);
    }
  }
#pragma unroll
  for (int dt = 0; dt < 4; ++dt)
#pragma unroll
    for (int r = 0; r < 8; ++r) {
      int nrow = rt * 16 + r + 8 * kh;
      whp[((size_t)h * NN + nrow) * FHID + dt * 16 + m] = (bf16_t)acc[dt][r];
    }
}

// ---------------- swizzle Wh into WMMA B-operand lane layout ----------------
// whB[h][jt(128)][dt(4)][lane(32)][t(16)] = Wh[h][jt*32+(lane/16)*16+t][dt*16+lane%16]
__global__ void k_swz(const bf16_t* __restrict__ whp, bf16_t* __restrict__ whB) {
  int idx  = blockIdx.x * blockDim.x + threadIdx.x;
  int lane = idx & 31;
  int dt   = (idx >> 5) & 3;
  int jt   = (idx >> 7) & 127;
  int h    = idx >> 14;
  if (h >= HH) return;
  bf16_t* out = whB + (size_t)idx * 16;
  int d  = dt * 16 + (lane & 15);
  int jb = jt * 32 + (lane >> 4) * 16;
#pragma unroll
  for (int t = 0; t < 16; ++t)
    out[t] = whp[((size_t)h * NN + jb + t) * FHID + d];
}

// ---------------- e_src / e_dst per (head, node) ----------------
__global__ void k_esed(const bf16_t* __restrict__ whp, const float* __restrict__ a_heads,
                       float* __restrict__ es, float* __restrict__ ed,
                       float* __restrict__ edT) {
  int idx = blockIdx.x * blockDim.x + threadIdx.x;   // HH*NN
  int h = idx / NN, n = idx % NN;
  const bf16_t* w  = whp + ((size_t)h * NN + n) * FHID;
  const float*  as = a_heads + h * 2 * FHID;
  float s = 0.f, d = 0.f;
#pragma unroll 8
  for (int k = 0; k < FHID; ++k) {
    float v = (float)w[k];
    s += v * as[k];
    d += v * as[FHID + k];
  }
  es[h * NN + n] = s;
  ed[h * NN + n] = d;
  edT[n * HH + h] = d;
}

// ---------------- pass 1: per-row neighbor max of ed, all heads, via bitmask ----------------
__global__ void k_rowmax(const unsigned* __restrict__ mask, const float* __restrict__ edT,
                         float* __restrict__ Mmax) {
  const int i = blockIdx.x, tid = threadIdx.x;   // 128 threads: one mask word each
  float mx[HH];
#pragma unroll
  for (int h = 0; h < HH; ++h) mx[h] = -3.0e38f;
  unsigned w = mask[(size_t)i * NW + tid];
  while (w) {
    int b = __ffs(w) - 1;
    w &= w - 1;
    const float* z = edT + (size_t)(tid * 32 + b) * HH;
#pragma unroll
    for (int h = 0; h < HH; ++h) mx[h] = fmaxf(mx[h], z[h]);
  }
#pragma unroll
  for (int h = 0; h < HH; ++h)
    for (int msk = 16; msk; msk >>= 1) mx[h] = fmaxf(mx[h], __shfl_xor(mx[h], msk, 32));
  __shared__ float red[4][HH];
  if ((tid & 31) == 0)
#pragma unroll
    for (int h = 0; h < HH; ++h) red[tid >> 5][h] = mx[h];
  __syncthreads();
  if (tid < HH) {
    float v = fmaxf(fmaxf(red[0][tid], red[1][tid]), fmaxf(red[2][tid], red[3][tid]));
    Mmax[i * HH + tid] = v;
  }
}

// ---------------- pass 2 (hot): fused masked-softmax + P~ @ Wh via WMMA ----------------
// block = 8 waves = 8 row-tiles of the SAME head -> whB tile shared (L1 reuse).
// A operand built entirely in registers (bitmask + readlane), no LDS.
__global__ void __launch_bounds__(256) k_attn(
    const unsigned* __restrict__ mask, const float* __restrict__ es,
    const float* __restrict__ ed, const float* __restrict__ Mmax,
    const bf16_t* __restrict__ whB, float* __restrict__ hbuf) {
  const int lane = threadIdx.x & 31;
  const int wid  = threadIdx.x >> 5;
  const int h    = blockIdx.y;
  const int rt   = blockIdx.x * 8 + wid;
  const int m    = lane & 15;
  const int kh   = lane >> 4;

  const int   myrow = rt * 16 + m;            // lane owns this output row
  const float esv   = es[h * NN + myrow];
  const float mv    = lrelu(esv + Mmax[myrow * HH + h]);  // lrelu monotone => row max
  float       ssum  = 0.f;

  v8f acc[4];
#pragma unroll
  for (int dt = 0; dt < 4; ++dt)
#pragma unroll
    for (int q = 0; q < 8; ++q) acc[dt][q] = 0.f;

  const float*    edh  = ed + (size_t)h * NN;
  const unsigned* mrow = mask + (size_t)myrow * NW;
  const bf16_t*   bh   = whB + (size_t)h * 128 * 2048;

  for (int jt = 0; jt < NW; ++jt) {
    const unsigned wm  = mrow[jt];            // this lane's row mask word
    const float    edv = edh[jt * 32 + lane]; // column value, lane = column

    if (jt + 1 < NW)  // prefetch next B tile (global_prefetch_b8), 4KB across 32 lanes
      __builtin_prefetch((const char*)(bh + (size_t)(jt + 1) * 2048) + lane * 128, 0, 3);

    // Build A operand (16x32 p~ tile) directly in WMMA lane layout:
    // element i -> column c = kh*16 + i, row = lane&15.
    v16bf a;
#pragma unroll
    for (int i = 0; i < 16; ++i) {
      const int c0  = i;                      // kh==0 column (literal)
      const int c1  = c0 + 16;                // kh==1 column (literal)
      float     e0  = rdlane(edv, c0);
      float     e1  = rdlane(edv, c1);
      float     edc = kh ? e1 : e0;
      unsigned  bit = kh ? ((wm >> c1) & 1u) : ((wm >> c0) & 1u);
      float     p   = bit ? __expf(lrelu(esv + edc) - mv) : 0.f;
      ssum += p;
      a[i] = (bf16_t)p;
    }

    const v16bf* bp = (const v16bf*)(bh + (size_t)jt * 2048);
#pragma unroll
    for (int dt = 0; dt < 4; ++dt) {
      v16bf b = bp[dt * 32 + lane];           // coalesced 32B/lane
      acc[dt] = __builtin_amdgcn_wmma_f32_16x16x32_bf16(false, a, false, b,
                                                        (short)0, acc[dt], false, false);
    }
  }

  // lane l and l+16 hold the two column-halves of row (l&15): one xor-swap adds them
  ssum += __shfl_xor(ssum, 16, 32);

  float inv[8];
#pragma unroll
  for (int r = 0; r < 8; ++r) {
    float s0 = rdlane(ssum, r);       // row r total (lives in lane r)
    float s1 = rdlane(ssum, r + 8);   // row r+8 total
    inv[r] = 1.f / (kh ? s1 : s0);
  }
#pragma unroll
  for (int dt = 0; dt < 4; ++dt)
#pragma unroll
    for (int r = 0; r < 8; ++r) {
      int   row = rt * 16 + r + 8 * kh;
      float v   = acc[dt][r] * inv[r];
      v = v > 0.f ? v : expm1f(v);    // ELU
      hbuf[(size_t)row * (HH * FHID) + h * FHID + dt * 16 + m] = v;
    }
}

// ---------------- Wh2 = h @ W_out (4096 x 512 . 512) ----------------
__global__ void k_w2(const float* __restrict__ hbuf, const float* __restrict__ W_out,
                     float* __restrict__ w2) {
  int lane = threadIdx.x & 31, wid = threadIdx.x >> 5;
  int row  = blockIdx.x * 8 + wid;
  const float* hr = hbuf + (size_t)row * (HH * FHID);
  float s = 0.f;
  for (int k = lane; k < HH * FHID; k += 32) s += hr[k] * W_out[k];
  for (int msk = 16; msk; msk >>= 1) s += __shfl_xor(s, msk, 32);
  if (lane == 0) w2[row] = s;
}

// ---------------- stage-2 pass 1: per-row neighbor max of a1*w2[j] ----------------
__global__ void k_rowmax2(const unsigned* __restrict__ mask, const float* __restrict__ w2,
                          const float* __restrict__ a_out, float* __restrict__ m2) {
  int   i = blockIdx.x, tid = threadIdx.x;   // 128 threads
  float a1 = a_out[1];
  float z  = -3.0e38f;
  unsigned w = mask[(size_t)i * NW + tid];
  while (w) {
    int b = __ffs(w) - 1;
    w &= w - 1;
    z = fmaxf(z, a1 * w2[tid * 32 + b]);
  }
  for (int msk = 16; msk; msk >>= 1) z = fmaxf(z, __shfl_xor(z, msk, 32));
  __shared__ float red[4];
  if ((tid & 31) == 0) red[tid >> 5] = z;
  __syncthreads();
  if (tid == 0) {
    float v = fmaxf(fmaxf(red[0], red[1]), fmaxf(red[2], red[3]));
    m2[i] = lrelu(a_out[0] * w2[i] + v);
  }
}

// ---------------- stage-2 pass 2 + ELU + log_softmax(singleton) ----------------
__global__ void k_final(const unsigned* __restrict__ mask, const float* __restrict__ w2,
                        const float* __restrict__ a_out, const float* __restrict__ m2,
                        float* __restrict__ out) {
  int   i  = blockIdx.x, tid = threadIdx.x;  // 128 threads
  float a0 = a_out[0], a1 = a_out[1];
  float w2i = w2[i], mi = m2[i];
  float s = 0.f, ac = 0.f;
  unsigned w = mask[(size_t)i * NW + tid];
  while (w) {
    int b = __ffs(w) - 1;
    w &= w - 1;
    float wj = w2[tid * 32 + b];
    float p  = __expf(lrelu(a0 * w2i + a1 * wj) - mi);
    s += p;
    ac += p * wj;
  }
  for (int msk = 16; msk; msk >>= 1) {
    s += __shfl_xor(s, msk, 32);
    ac += __shfl_xor(ac, msk, 32);
  }
  __shared__ float rs[4], ra[4];
  if ((tid & 31) == 0) { rs[tid >> 5] = s; ra[tid >> 5] = ac; }
  __syncthreads();
  if (tid == 0) {
    float S = rs[0] + rs[1] + rs[2] + rs[3];
    float A = ra[0] + ra[1] + ra[2] + ra[3];
    float h2 = A / S;
    float v  = h2 > 0.f ? h2 : expm1f(h2);
    // log_softmax over axis of size 1: v - logsumexp(v) == 0 exactly
    out[i] = v - v;
  }
}

extern "C" void kernel_launch(void* const* d_in, const int* in_sizes, int n_in,
                              void* d_out, int out_size, void* d_ws, size_t ws_size,
                              hipStream_t stream) {
  (void)in_sizes; (void)n_in; (void)out_size; (void)ws_size;
  const float* x       = (const float*)d_in[0];
  const float* adj     = (const float*)d_in[1];
  const float* W_heads = (const float*)d_in[2];
  const float* a_heads = (const float*)d_in[3];
  const float* W_out   = (const float*)d_in[4];
  const float* a_out   = (const float*)d_in[5];
  float*       out     = (float*)d_out;

  char* p     = (char*)d_ws;
  auto  alloc = [&](size_t bytes) -> void* {
    void* r = (void*)p;
    p += (bytes + 255) & ~(size_t)255;
    return r;
  };
  unsigned* msk  = (unsigned*)alloc((size_t)NN * NW * 4);          // 2 MB bitmask
  bf16_t*   xbf  = (bf16_t*)alloc((size_t)NN * FIN * 2);
  bf16_t*   wbf  = (bf16_t*)alloc((size_t)HH * FIN * FHID * 2);
  bf16_t*   whp  = (bf16_t*)alloc((size_t)HH * NN * FHID * 2);
  bf16_t*   whB  = (bf16_t*)alloc((size_t)HH * NN * FHID * 2);
  float*    es   = (float*)alloc((size_t)HH * NN * 4);
  float*    ed   = (float*)alloc((size_t)HH * NN * 4);
  float*    edT  = (float*)alloc((size_t)NN * HH * 4);
  float*    Mmax = (float*)alloc((size_t)NN * HH * 4);
  float*    hbuf = (float*)alloc((size_t)NN * HH * FHID * 4);
  float*    w2   = (float*)alloc((size_t)NN * 4);
  float*    m2   = (float*)alloc((size_t)NN * 4);

  k_mask<<<(NN * NN) / 256, 256, 0, stream>>>(adj, msk);
  k_cvt_x<<<(NN * FIN + 255) / 256, 256, 0, stream>>>(x, xbf);
  k_cvt_w<<<(HH * FIN * FHID + 255) / 256, 256, 0, stream>>>(W_heads, wbf);
  k_wh<<<dim3(NN / 128, HH), 256, 0, stream>>>(xbf, wbf, whp);
  k_swz<<<(HH * 128 * 4 * 32) / 256, 256, 0, stream>>>(whp, whB);
  k_esed<<<(HH * NN) / 256, 256, 0, stream>>>(whp, a_heads, es, ed, edT);
  k_rowmax<<<NN, 128, 0, stream>>>(msk, edT, Mmax);
  k_attn<<<dim3(NN / 128, HH), 256, 0, stream>>>(msk, es, ed, Mmax, whB, hbuf);
  k_w2<<<NN / 8, 256, 0, stream>>>(hbuf, W_out, w2);
  k_rowmax2<<<NN, 128, 0, stream>>>(msk, w2, a_out, m2);
  k_final<<<NN, 128, 0, stream>>>(msk, w2, a_out, m2, out);
}
